// MambaSegDecoder_85882166051443
// MI455X (gfx1250) — matrix operand order repaired and checked by reading, not compiled, not run on hardware
//
#include <hip/hip_runtime.h>

// MambaSegDecoder forward for gfx1250 (wave32, WMMA + TDM async B-staging).

typedef __attribute__((ext_vector_type(16))) _Float16 v16h;
typedef __attribute__((ext_vector_type(8)))  float    v8f;
typedef __attribute__((ext_vector_type(4)))  unsigned int u32x4;
typedef __attribute__((ext_vector_type(8)))  int      i32x8;
typedef __attribute__((ext_vector_type(4)))  int      i32x4;

#define LN_EPS 1e-5f
#define BN_SCALEF 0.9999950000374997f   // 1/sqrt(1+1e-5)

#if __has_builtin(__builtin_amdgcn_tensor_load_to_lds) && __has_builtin(__builtin_amdgcn_s_wait_tensorcnt)
#define USE_TDM 1
#else
#define USE_TDM 0
#endif

__device__ __forceinline__ float siluf(float x) { return x / (1.f + expf(-x)); }

// ---------------------------------------------------------------------------
// Strided batched GEMM, f16 WMMA, f32 accumulate.
//   C[m,n] (+= bias) = sum_k A[m,k]*B[k,n]
// A element-k stride == 1, B element-n stride == 1 (true at every call site).
// Each wave computes a 64x16 strip: 4 M-tiles reuse one B fragment.
// Fast path: B panels DMA'd to LDS by the Tensor Data Mover (double-buffered,
// TENSORcnt-tracked); A via float4 global loads. Edge path: clamped loads.
// bias_mode: 0 none, 1 per-n, 2 per-m.
// ---------------------------------------------------------------------------
__global__ void gemm_wmma_f16(const float* __restrict__ A, const float* __restrict__ B,
                              float* __restrict__ Cp, int M, int N, int K,
                              int sAm, int sBk, int sCm, int sCn,
                              long aBS, long bBS, long cBS, int aMod,
                              const float* __restrict__ bias, int bias_mode) {
#if USE_TDM
  // 2 buffers x 8 waves x (32 rows x 17 floats padded) -- pad kills bank conflicts
  __shared__ float bstage[2][8][544];
#endif
  int bz = blockIdx.y;
  const float* Ab = A + (size_t)(aMod > 0 ? (bz % aMod) : bz) * aBS;
  const float* Bb = B + (size_t)bz * bBS;
  float* Cb = Cp + (size_t)bz * cBS;
  int stripsM = (M + 63) >> 6;
  int tilesN = (N + 15) >> 4;
  int numW = stripsM * tilesN;
  // force scalar: value is wave-invariant, readfirstlane makes it provably so
  int wave = __builtin_amdgcn_readfirstlane((int)((blockIdx.x * blockDim.x + threadIdx.x) >> 5));
  if (wave >= numW) return;
  int tm0 = (wave / tilesN) << 6;     // scalar
  int tn = (wave % tilesN) << 4;      // scalar
  int lane = (int)(threadIdx.x & 31);
  int half = lane >> 4;               // 0: lanes 0-15, 1: lanes 16-31
  int l16 = lane & 15;
  int n = tn + l16;                   // B/C column (vector)
  int rem = M - tm0;                  // scalar

  v8f zero8 = {0.f,0.f,0.f,0.f,0.f,0.f,0.f,0.f};
  v8f acc[4] = {zero8, zero8, zero8, zero8};

  bool fast = ((K & 31) == 0) && (tn + 16 <= N);   // scalar
  if (fast) {
#if USE_TDM
    int wib = __builtin_amdgcn_readfirstlane((int)(threadIdx.x >> 5) & 7);
    float* buf0 = &bstage[0][wib][0];
    float* buf1 = &bstage[1][wib][0];
    // LDS aperture: low 32 bits of the flat pointer are the LDS byte offset
    unsigned lds0 = (unsigned)(size_t)(void*)buf0;
    unsigned lds1 = (unsigned)(size_t)(void*)buf1;
    // D# group1: data_size=4B, pad 1 dword per 16-dword row, tensor 16x32,
    // tile 16x32, dim0_stride = sBk (elements)
    i32x8 g1;
    g1[0] = (int)((2u << 16) | (1u << 20) | (3u << 22));  // data_size|pad_en|pad_interval=16dw
    g1[1] = (int)(16u << 16);   // tensor_dim0 = 16 (low bits at dword bit 48)
    g1[2] = (int)(32u << 16);   // tensor_dim1 = 32
    g1[3] = (int)(16u << 16);   // tile_dim0 = 16
    g1[4] = 32;                 // tile_dim1 = 32
    g1[5] = sBk;                // tensor_dim0_stride (elements)
    g1[6] = 0; g1[7] = 0;
    i32x4 gz; gz[0] = 0; gz[1] = 0; gz[2] = 0; gz[3] = 0;
    i32x8 gz8; gz8[0]=0; gz8[1]=0; gz8[2]=0; gz8[3]=0; gz8[4]=0; gz8[5]=0; gz8[6]=0; gz8[7]=0;
    unsigned long long gaddr = (unsigned long long)(size_t)(Bb + tn);
    unsigned long long cstep = (unsigned long long)((unsigned)sBk) * 128ull; // 32 rows * 4B
    auto issue = [&](int chunk, unsigned lds) {
      unsigned long long a = gaddr + (unsigned long long)chunk * cstep;
      u32x4 g0;
      g0[0] = 1u;                                    // count=1, user mode
      g0[1] = lds;                                   // lds_addr (bytes)
      g0[2] = (unsigned)(a & 0xFFFFFFFFull);         // global_addr[31:0]
      g0[3] = (unsigned)((a >> 32) & 0x1FFFFFFull) | (2u << 30);  // addr[56:32], type=2
      __builtin_amdgcn_tensor_load_to_lds(g0, g1, gz, gz, gz8, 0);
    };
    issue(0, lds0);
    for (int k0 = 0; k0 < K; k0 += 32) {
      int j = k0 >> 5;                      // chunk index (scalar)
      if (k0 + 32 < K) {
        issue(j + 1, (j & 1) ? lds0 : lds1);
        __builtin_amdgcn_s_wait_tensorcnt(1);   // chunk j has landed
      } else {
        __builtin_amdgcn_s_wait_tensorcnt(0);
      }
      const float* bp = (j & 1) ? buf1 : buf0;
      v16h bf;
#pragma unroll
      for (int i = 0; i < 16; ++i)            // row k_local, padded stride 17
        bf[i] = (_Float16)bp[(half * 8 + (i & 7) + ((i >> 3) << 4)) * 17 + l16];
#else
    for (int k0 = 0; k0 < K; k0 += 32) {
      v16h bf;
      int bb0 = (k0 + half * 8) * sBk + n;
#pragma unroll
      for (int i = 0; i < 16; ++i)   // B 32x16: lanes 0-15 K=0..7,16..23; 16-31 K=8..15,24..31
        bf[i] = (_Float16)Bb[bb0 + ((i & 7) + ((i >> 3) << 4)) * sBk];
#endif
#pragma unroll
      for (int t = 0; t < 4; ++t) {
        if (t * 16 >= rem) break;    // scalar branch
        int m = tm0 + t * 16 + l16;
        v16h af;
        if (tm0 + t * 16 + 16 <= M) {          // scalar branch
          const float* ap = Ab + m * sAm + k0 + half * 8;
          float4 a0 = *(const float4*)(ap);
          float4 a1 = *(const float4*)(ap + 4);
          float4 a2 = *(const float4*)(ap + 16);
          float4 a3 = *(const float4*)(ap + 20);
          af[0]=(_Float16)a0.x; af[1]=(_Float16)a0.y; af[2]=(_Float16)a0.z; af[3]=(_Float16)a0.w;
          af[4]=(_Float16)a1.x; af[5]=(_Float16)a1.y; af[6]=(_Float16)a1.z; af[7]=(_Float16)a1.w;
          af[8]=(_Float16)a2.x; af[9]=(_Float16)a2.y; af[10]=(_Float16)a2.z; af[11]=(_Float16)a2.w;
          af[12]=(_Float16)a3.x; af[13]=(_Float16)a3.y; af[14]=(_Float16)a3.z; af[15]=(_Float16)a3.w;
          if (k0 + 32 < K) __builtin_prefetch(ap + 32, 0, 0);  // global_prefetch_b8
        } else {
          // partial last tile: clamp row, garbage rows discarded at store
          int mc = m < M ? m : M - 1;
          const float* ap = Ab + mc * sAm + k0 + half * 8;
#pragma unroll
          for (int i = 0; i < 8; ++i) {
            af[i]     = (_Float16)ap[i];
            af[8 + i] = (_Float16)ap[16 + i];
          }
        }
        acc[t] = __builtin_amdgcn_wmma_f32_16x16x32_f16(false, af, false, bf, (short)0,
                                                        acc[t], false, false);
      }
    }
  } else {
    // ---- edge path: clamped addresses (always legal) + selects, no branches ----
    int nc = n < N ? n : N - 1;
    for (int k0 = 0; k0 < K; k0 += 32) {
      v16h bf;
#pragma unroll
      for (int i = 0; i < 16; ++i) {
        int kk = k0 + half * 8 + (i & 7) + ((i >> 3) << 4);
        int kc = kk < K ? kk : K - 1;
        float v = Bb[kc * sBk + nc];
        bf[i] = (kk < K) ? (_Float16)v : (_Float16)0.f;  // zero B kills k>=K products
      }
#pragma unroll
      for (int t = 0; t < 4; ++t) {
        if (t * 16 >= rem) break;    // scalar branch
        int m = tm0 + t * 16 + l16;
        int mc = m < M ? m : M - 1;
        const float* ap = Ab + mc * sAm;
        v16h af;
#pragma unroll
        for (int i = 0; i < 8; ++i) {
          int ka = k0 + half * 8 + i;
          int kb = ka + 16;
          af[i]     = (_Float16)ap[ka < K ? ka : K - 1];
          af[8 + i] = (_Float16)ap[kb < K ? kb : K - 1];
        }
        acc[t] = __builtin_amdgcn_wmma_f32_16x16x32_f16(false, af, false, bf, (short)0,
                                                        acc[t], false, false);
      }
    }
  }

#pragma unroll
  for (int t = 0; t < 4; ++t) {
    if (t * 16 >= rem) break;        // scalar branch
#pragma unroll
    for (int r = 0; r < 8; ++r) {
      int mr = tm0 + t * 16 + r + half * 8;   // C: VGPR r -> row r (+8 for high lanes)
      if (mr < M && n < N) {
        float v = acc[t][r];
        if (bias_mode == 1) v += bias[n];
        else if (bias_mode == 2) v += bias[mr];
        Cb[mr * sCm + n * sCn] = v;
      }
    }
  }
}

// ---------------------------------------------------------------------------
// LayerNorm over last dim (token-major [Ntok, C]); one wave32 per token.
// Optional gate: out *= silu(gate[tok*gstride + c]).
// ---------------------------------------------------------------------------
__global__ void ln_gate_kernel(const float* __restrict__ x, float* __restrict__ out,
                               const float* __restrict__ g, const float* __restrict__ b,
                               const float* __restrict__ gate, long gstride,
                               long Ntok, int C) {
  long wid = (long)(blockIdx.x * blockDim.x + threadIdx.x) >> 5;
  int lane = threadIdx.x & 31;
  if (wid >= Ntok) return;
  const float* xr = x + wid * C;
  float s = 0.f, s2 = 0.f;
  for (int c = lane; c < C; c += 32) { float v = xr[c]; s += v; s2 += v * v; }
  for (int o = 16; o > 0; o >>= 1) { s += __shfl_xor(s, o, 32); s2 += __shfl_xor(s2, o, 32); }
  float mean = s / C;
  float var = s2 / C - mean * mean;
  float rstd = rsqrtf(var + LN_EPS);
  float* orow = out + wid * C;
  for (int c = lane; c < C; c += 32) {
    float v = (xr[c] - mean) * rstd * g[c] + b[c];
    if (gate) v *= siluf(gate[wid * gstride + c]);
    orow[c] = v;
  }
}

// NCHW -> token-major [b, l, c]
__global__ void nchw_to_tok(const float* __restrict__ src, float* __restrict__ dst,
                            int Bn, int C, long L) {
  long idx = (long)blockIdx.x * blockDim.x + threadIdx.x;
  long total = (long)Bn * C * L;
  if (idx >= total) return;
  long l = idx % L; int c = (int)((idx / L) % C); int b = (int)(idx / (L * C));
  dst[((long)b * L + l) * C + c] = src[idx];
}

// token-major -> NCHW
__global__ void tok_to_nchw(const float* __restrict__ src, float* __restrict__ dst,
                            int Bn, int C, long L) {
  long idx = (long)blockIdx.x * blockDim.x + threadIdx.x;
  long total = (long)Bn * C * L;
  if (idx >= total) return;
  long l = idx % L; int c = (int)((idx / L) % C); int b = (int)(idx / (L * C));
  dst[idx] = src[((long)b * L + l) * C + c];
}

// cat[tok, 0:C]=xexp tokens, cat[tok, C:2C]=skip (NCHW, transposed on the fly)
__global__ void concat_build(const float* __restrict__ xexp, const float* __restrict__ skip,
                             float* __restrict__ cat, int Bn, long L, int C) {
  long idx = (long)blockIdx.x * blockDim.x + threadIdx.x;
  long total = (long)Bn * L * 2 * C;
  if (idx >= total) return;
  int c2 = (int)(idx % (2 * C));
  long tok = idx / (2 * C);
  int b = (int)(tok / L); long l = tok % L;
  cat[idx] = (c2 < C) ? xexp[tok * C + c2]
                      : skip[((long)b * C + (c2 - C)) * L + l];
}

// pixel shuffle: in token-major [b, Hin*Win, s*s*c] -> out token-major [b, Hin*s*Win*s, c]
__global__ void pixel_shuffle_k(const float* __restrict__ in, float* __restrict__ out,
                                int Bn, int Hin, int Win, int s, int c) {
  long idx = (long)blockIdx.x * blockDim.x + threadIdx.x;
  long total = (long)Bn * Hin * s * Win * s * c;
  if (idx >= total) return;
  int cc = (int)(idx % c);
  long t = idx / c;
  int wo = (int)(t % (Win * s)); t /= (Win * s);
  int ho = (int)(t % (Hin * s));
  int b = (int)(t / (Hin * s));
  int hi = ho / s, s1 = ho % s, wi = wo / s, s2 = wo % s;
  out[idx] = in[((long)(b * Hin + hi) * Win + wi) * (long)(s * s * c) + (long)(s1 * s + s2) * c + cc];
}

// depthwise 3x3 conv + bias + silu on xc (cols 0..D2-1 of xz, token-major),
// scattering into all four scan directions of xs [B,4,D2,L].
__global__ void dwconv_silu_xs(const float* __restrict__ xz, const float* __restrict__ w,
                               const float* __restrict__ bias, float* __restrict__ xs,
                               int Bn, int D2, int H, int W) {
  long idx = (long)blockIdx.x * blockDim.x + threadIdx.x;
  long L = (long)H * W;
  long total = (long)Bn * D2 * L;
  if (idx >= total) return;
  int l = (int)(idx % L);
  long t = idx / L;
  int d = (int)(t % D2); int b = (int)(t / D2);
  int h = l / W, wx = l % W;
  int stride = 2 * D2;
  const float* wd = w + (long)d * 9;
  float acc = bias[d];
#pragma unroll
  for (int dy = -1; dy <= 1; ++dy)
#pragma unroll
    for (int dx = -1; dx <= 1; ++dx) {
      int hh = h + dy, ww = wx + dx;
      if (hh >= 0 && hh < H && ww >= 0 && ww < W)
        acc += wd[(dy + 1) * 3 + (dx + 1)] * xz[((long)b * L + hh * W + ww) * stride + d];
    }
  float v = siluf(acc);
  long kst = (long)D2 * L;
  long base = (long)b * 4 * kst + (long)d * L;
  int l1 = wx * H + h;
  xs[base + l] = v;                    // k0: hw order
  xs[base + kst + l1] = v;             // k1: wh order
  xs[base + 2 * kst + (L - 1 - l)] = v;  // k2: reversed hw
  xs[base + 3 * kst + (L - 1 - l1)] = v; // k3: reversed wh
}

// delta = softplus(delta + dtproj_b[k,d])
__global__ void softplus_bias(float* __restrict__ delta, const float* __restrict__ dtb,
                              int Bn, int D2, long L) {
  long idx = (long)blockIdx.x * blockDim.x + threadIdx.x;
  long total = (long)Bn * 4 * D2 * L;
  if (idx >= total) return;
  int d = (int)((idx / L) % D2);
  int k = (int)((idx / (L * D2)) % 4);
  float v = delta[idx] + dtb[k * D2 + d];
  delta[idx] = (v > 20.f) ? v : log1pf(expf(v));
}

// selective scan: one thread per (b,k,d), NS=16 states in registers.
__global__ void scan_kernel(const float* __restrict__ xs, const float* __restrict__ delta,
                            const float* __restrict__ dbl, const float* __restrict__ A_log,
                            const float* __restrict__ Dp, float* __restrict__ oy,
                            int Bn, int D2, long L, int R) {
  long idx = (long)blockIdx.x * blockDim.x + threadIdx.x;
  long total = (long)Bn * 4 * D2;
  if (idx >= total) return;
  int d = (int)(idx % D2);
  int k = (int)((idx / D2) % 4);
  int b = (int)(idx / ((long)D2 * 4));
  float An[16];
#pragma unroll
  for (int n = 0; n < 16; ++n) An[n] = -expf(A_log[((long)k * D2 + d) * 16 + n]);
  float Dv = Dp[k * D2 + d];
  long chan = ((long)(b * 4 + k) * D2 + d) * L;
  const float* u = xs + chan;
  const float* dl = delta + chan;
  const float* Bm = dbl + (long)(b * 4 + k) * (R + 32) * L + (long)R * L;
  const float* Cm = Bm + 16 * L;
  float* o = oy + chan;
  float h[16];
#pragma unroll
  for (int n = 0; n < 16; ++n) h[n] = 0.f;
  for (long l = 0; l < L; ++l) {
    float dt = dl[l], uu = u[l], du = dt * uu;
    float y = 0.f;
#pragma unroll
    for (int n = 0; n < 16; ++n) {
      h[n] = expf(dt * An[n]) * h[n] + du * Bm[(long)n * L + l];
      y += h[n] * Cm[(long)n * L + l];
    }
    o[l] = y + Dv * uu;
  }
}

// combine four scan directions -> token-major [Ntok, D2]
__global__ void combine_kernel(const float* __restrict__ oy, float* __restrict__ out,
                               int Bn, int D2, int H, int W) {
  long idx = (long)blockIdx.x * blockDim.x + threadIdx.x;
  long L = (long)H * W;
  long total = (long)Bn * D2 * L;
  if (idx >= total) return;
  int l = (int)(idx % L);
  int d = (int)((idx / L) % D2);
  int b = (int)(idx / (L * D2));
  long kst = (long)D2 * L;
  long base = (long)b * 4 * kst + (long)d * L;
  int h = l / W, wx = l % W;
  int l1 = wx * H + h;
  float v = oy[base + l] + oy[base + 2 * kst + (L - 1 - l)] +
            oy[base + kst + l1] + oy[base + 3 * kst + (L - 1 - l1)];
  out[((long)b * L + l) * D2 + d] = v;
}

__global__ void add_inplace(float* __restrict__ x, const float* __restrict__ t, long n) {
  long i = (long)blockIdx.x * blockDim.x + threadIdx.x;
  if (i < n) x[i] += t[i];
}
__global__ void add2(float* __restrict__ o, const float* __restrict__ a,
                     const float* __restrict__ b, long n) {
  long i = (long)blockIdx.x * blockDim.x + threadIdx.x;
  if (i < n) o[i] = a[i] + b[i];
}
__global__ void copy_kernel(float* __restrict__ d, const float* __restrict__ s, long n) {
  long i = (long)blockIdx.x * blockDim.x + threadIdx.x;
  if (i < n) d[i] = s[i];
}

// AFR: per-(b,c) mean & max over tokens (token-major x)
__global__ void afr_reduce(const float* __restrict__ x, float* __restrict__ avg,
                           float* __restrict__ mx, int Bn, int C, long L) {
  int bc = blockIdx.x;
  int b = bc / C, c = bc % C;
  int tid = threadIdx.x;
  float s = 0.f, m = -3.4e38f;
  for (long l = tid; l < L; l += blockDim.x) {
    float v = x[((long)b * L + l) * C + c];
    s += v; m = fmaxf(m, v);
  }
  __shared__ float ss[256], sm[256];
  ss[tid] = s; sm[tid] = m;
  __syncthreads();
  for (int st = 128; st > 0; st >>= 1) {
    if (tid < st) { ss[tid] += ss[tid + st]; sm[tid] = fmaxf(sm[tid], sm[tid + st]); }
    __syncthreads();
  }
  if (tid == 0) { avg[bc] = ss[0] / (float)L; mx[bc] = sm[0]; }
}

// att = sigmoid( relu(avg@W1)@W2 + relu(mx@W1)@W2 )
__global__ void afr_att(const float* __restrict__ avg, const float* __restrict__ mx,
                        const float* __restrict__ W1, const float* __restrict__ W2,
                        float* __restrict__ att, int C, int red) {
  int b = blockIdx.x;
  int tid = threadIdx.x;
  __shared__ float hsum[32];
  if (tid < red) {
    float sa = 0.f, sm = 0.f;
    for (int c = 0; c < C; ++c) {
      float w = W1[(long)c * red + tid];
      sa += avg[b * C + c] * w;
      sm += mx[b * C + c] * w;
    }
    hsum[tid] = fmaxf(sa, 0.f) + fmaxf(sm, 0.f);
  }
  __syncthreads();
  for (int c = tid; c < C; c += blockDim.x) {
    float s = 0.f;
    for (int r = 0; r < red; ++r) s += hsum[r] * W2[(long)r * C + c];
    att[b * C + c] = 1.f / (1.f + expf(-s));
  }
}

__global__ void afr_scale(float* __restrict__ x, const float* __restrict__ att,
                          int Bn, int C, long L) {
  long idx = (long)blockIdx.x * blockDim.x + threadIdx.x;
  long total = (long)Bn * L * C;
  if (idx >= total) return;
  int c = (int)(idx % C);
  long tok = idx / C;
  int b = (int)(tok / L);
  x[idx] *= att[b * C + c];
}

// BAM edge: depthwise laplacian [[1,1,1],[1,-8,1],[1,1,1]], pad 1, NCHW
__global__ void laplacian3x3(const float* __restrict__ in, float* __restrict__ out,
                             int Bn, int C, int H, int W) {
  long idx = (long)blockIdx.x * blockDim.x + threadIdx.x;
  long HW = (long)H * W;
  long total = (long)Bn * C * HW;
  if (idx >= total) return;
  int wx = (int)(idx % W);
  int h = (int)((idx / W) % H);
  long bc = idx / HW;
  const float* p = in + bc * HW;
  float acc = -8.f * p[(long)h * W + wx];
#pragma unroll
  for (int dy = -1; dy <= 1; ++dy)
#pragma unroll
    for (int dx = -1; dx <= 1; ++dx) {
      if (dy == 0 && dx == 0) continue;
      int hh = h + dy, ww = wx + dx;
      if (hh >= 0 && hh < H && ww >= 0 && ww < W) acc += p[(long)hh * W + ww];
    }
  out[idx] = acc;
}

// relu(x * g[c]*BN_SCALE + b[c]) on a channel sub-range of an NCHW buffer
__global__ void bnrelu(float* __restrict__ data, const float* __restrict__ g,
                       const float* __restrict__ b, int Bn, int Ctot, int coff,
                       int Cs, long L) {
  long idx = (long)blockIdx.x * blockDim.x + threadIdx.x;
  long total = (long)Bn * Cs * L;
  if (idx >= total) return;
  long l = idx % L;
  int c = (int)((idx / L) % Cs);
  int bq = (int)(idx / (L * Cs));
  long p = ((long)bq * Ctot + coff + c) * L + l;
  float v = data[p] * (g[c] * BN_SCALEF) + b[c];
  data[p] = fmaxf(v, 0.f);
}

// 3x3 conv (pad 1, no conv bias) + BN + relu, NCHW direct
__global__ void conv3x3_bnrelu(const float* __restrict__ in, const float* __restrict__ w,
                               const float* __restrict__ g, const float* __restrict__ bb,
                               float* __restrict__ out, int Bn, int Cin, int Cout,
                               int H, int W) {
  long idx = (long)blockIdx.x * blockDim.x + threadIdx.x;
  long HW = (long)H * W;
  long total = (long)Bn * Cout * HW;
  if (idx >= total) return;
  int wx = (int)(idx % W);
  int h = (int)((idx / W) % H);
  int o = (int)((idx / HW) % Cout);
  int b = (int)(idx / (HW * Cout));
  float acc = 0.f;
  for (int c = 0; c < Cin; ++c) {
    const float* p = in + ((long)b * Cin + c) * HW;
    const float* wk = w + ((long)o * Cin + c) * 9;
#pragma unroll
    for (int dy = -1; dy <= 1; ++dy)
#pragma unroll
      for (int dx = -1; dx <= 1; ++dx) {
        int hh = h + dy, ww = wx + dx;
        if (hh >= 0 && hh < H && ww >= 0 && ww < W)
          acc += p[(long)hh * W + ww] * wk[(dy + 1) * 3 + (dx + 1)];
      }
  }
  float v = acc * (g[o] * BN_SCALEF) + bb[o];
  out[idx] = fmaxf(v, 0.f);
}

// ---------------------------------------------------------------------------
extern "C" void kernel_launch(void* const* d_in, const int* in_sizes, int n_in,
                              void* d_out, int out_size, void* d_ws, size_t ws_size,
                              hipStream_t stream) {
  (void)in_sizes; (void)n_in; (void)out_size; (void)ws_size;
  auto P = [&](int i) { return (const float*)d_in[i]; };
  float* ws = (float*)d_ws;
  size_t off = 0;
  auto alloc = [&](size_t n) { float* p = ws + off; off += (n + 63) & ~(size_t)63; return p; };

  const int Bn = 4;
  dim3 blk(256);
  auto gridE = [&](long total) { return dim3((unsigned)((total + 255) / 256)); };

  float* lres = alloc((size_t)1 << 20);   // persistent stage output (token-major)
  float* avgb = alloc(Bn * 256);
  float* mxb  = alloc(Bn * 256);
  float* attb = alloc(Bn * 256);
  size_t arena = off;

  auto gemm = [&](const float* A, const float* Bp, float* Cp, int M, int N, int Kd,
                  int sAm, int sBk, int sCm, int sCn,
                  long aBS, long bBS, long cBS, int aMod, int batch,
                  const float* bias, int bias_mode) {
    int waves = ((M + 63) / 64) * ((N + 15) / 16);
    dim3 g((unsigned)((waves + 7) / 8), (unsigned)batch);
    gemm_wmma_f16<<<g, blk, 0, stream>>>(A, Bp, Cp, M, N, Kd, sAm, sBk, sCm, sCn,
                                         aBS, bBS, cBS, aMod, bias, bias_mode);
  };
  auto lnrun = [&](const float* x, float* o, const float* g, const float* b,
                   const float* gate, long gstride, long Ntok, int C) {
    ln_gate_kernel<<<gridE(Ntok * 32), blk, 0, stream>>>(x, o, g, b, gate, gstride, Ntok, C);
  };

  // lres <- tokens of skip3 (4,512,8,8)
  nchw_to_tok<<<gridE((long)Bn * 512 * 64), blk, 0, stream>>>(P(3), lres, Bn, 512, 64);

  const int HinA[3] = {8, 16, 32};
  const int CA[3]   = {256, 128, 64};
  const int DbA[3]  = {512, 256, 128};

  for (int s = 0; s < 3; ++s) {
    off = arena;
    int sb = 6 + 32 * s;
    int Hin = HinA[s], H = Hin * 2, W = H;
    long L = (long)H * W;
    int C = CA[s], Db = DbA[s], D2 = 2 * C, R = C / 16, red = C / 16;
    long NtokIn = (long)Bn * Hin * Hin;
    long Ntok = (long)Bn * L;

    float* T1    = alloc(NtokIn * 2 * Db);
    float* xexp  = alloc(Ntok * C);
    float* cat   = alloc(Ntok * 2 * C);
    float* x     = alloc(Ntok * C);
    float* xln   = alloc(Ntok * C);
    float* xz    = alloc(Ntok * 4 * C);
    float* xs    = alloc((size_t)Bn * 4 * D2 * L);
    float* dbl   = alloc((size_t)Bn * 4 * (R + 32) * L);
    float* delta = alloc((size_t)Bn * 4 * D2 * L);
    float* oy    = alloc((size_t)Bn * 4 * D2 * L);
    float* yc    = alloc(Ntok * D2);
    float* yln   = alloc(Ntok * D2);
    float* tmp   = alloc(Ntok * C);

    // patch expand (s=2) + LN
    gemm(lres, P(sb + 0), T1, (int)NtokIn, 2 * Db, Db, Db, 2 * Db, 2 * Db, 1,
         0, 0, 0, 0, 1, nullptr, 0);
    pixel_shuffle_k<<<gridE(Ntok * C), blk, 0, stream>>>(T1, xexp, Bn, Hin, Hin, 2, C);
    lnrun(xexp, xexp, P(sb + 1), P(sb + 2), nullptr, 0, Ntok, C);

    // concat with skip[2-s] + projection
    concat_build<<<gridE(Ntok * 2 * C), blk, 0, stream>>>(xexp, P(2 - s), cat, Bn, L, C);
    gemm(cat, P(sb + 3), x, (int)Ntok, C, 2 * C, 2 * C, C, C, 1,
         0, 0, 0, 0, 1, nullptr, 0);

    for (int j = 0; j < 2; ++j) {
      int bb = sb + 4 + 13 * j;
      lnrun(x, xln, P(bb + 0), P(bb + 1), nullptr, 0, Ntok, C);
      // in_W: [Ntok,C] @ [C,4C]
      gemm(xln, P(bb + 2), xz, (int)Ntok, 4 * C, C, C, 4 * C, 4 * C, 1,
           0, 0, 0, 0, 1, nullptr, 0);
      dwconv_silu_xs<<<gridE((long)Bn * D2 * L), blk, 0, stream>>>(
          xz, P(bb + 3), P(bb + 4), xs, Bn, D2, H, W);
      // xproj: per (b,k): [R+32, D2] @ [D2, L]
      gemm(P(bb + 5), xs, dbl, R + 32, (int)L, D2, D2, (int)L, (int)L, 1,
           (long)(R + 32) * D2, (long)D2 * L, (long)(R + 32) * L, 4, Bn * 4, nullptr, 0);
      // dtproj: per (b,k): [D2, R] @ [R, L]
      gemm(P(bb + 6), dbl, delta, D2, (int)L, R, R, (int)L, (int)L, 1,
           (long)D2 * R, (long)(R + 32) * L, (long)D2 * L, 4, Bn * 4, nullptr, 0);
      softplus_bias<<<gridE((long)Bn * 4 * D2 * L), blk, 0, stream>>>(delta, P(bb + 7), Bn, D2, L);
      scan_kernel<<<gridE((long)Bn * 4 * D2), blk, 0, stream>>>(
          xs, delta, dbl, P(bb + 8), P(bb + 9), oy, Bn, D2, L, R);
      combine_kernel<<<gridE((long)Bn * D2 * L), blk, 0, stream>>>(oy, yc, Bn, D2, H, W);
      // outnorm LN * silu(z); z = xz cols D2..2*D2
      lnrun(yc, yln, P(bb + 10), P(bb + 11), xz + D2, 4 * C, Ntok, D2);
      gemm(yln, P(bb + 12), tmp, (int)Ntok, C, D2, D2, C, C, 1,
           0, 0, 0, 0, 1, nullptr, 0);
      add_inplace<<<gridE(Ntok * C), blk, 0, stream>>>(x, tmp, Ntok * C);
    }

    // AFR
    afr_reduce<<<dim3((unsigned)(Bn * C)), blk, 0, stream>>>(x, avgb, mxb, Bn, C, L);
    afr_att<<<dim3((unsigned)Bn), blk, 0, stream>>>(avgb, mxb, P(sb + 30), P(sb + 31), attb, C, red);
    afr_scale<<<gridE(Ntok * C), blk, 0, stream>>>(x, attb, Bn, C, L);
    copy_kernel<<<gridE(Ntok * C), blk, 0, stream>>>(lres, x, Ntok * C);
  }

  // ---- final: expand s=4, LN, BAM, seg head ----
  off = arena;
  const long HW = 65536;            // 256x256
  float* T1f  = alloc((size_t)16777216);   // reused as XB after shuffle
  float* Xf   = alloc((size_t)16777216);   // reused as bam output
  float* edge = alloc((size_t)16777216);   // reused as fuse conv output
  float* fin  = alloc((size_t)16777216);   // 64-channel fuse input
  long NtokF = (long)Bn * 4096;

  gemm(lres, P(102), T1f, (int)NtokF, 1024, 64, 64, 1024, 1024, 1,
       0, 0, 0, 0, 1, nullptr, 0);
  pixel_shuffle_k<<<gridE((long)Bn * HW * 64), blk, 0, stream>>>(T1f, Xf, Bn, 64, 64, 4, 64);
  lnrun(Xf, Xf, P(103), P(104), nullptr, 0, (long)Bn * HW, 64);

  float* XB = T1f;
  tok_to_nchw<<<gridE((long)Bn * 64 * HW), blk, 0, stream>>>(Xf, XB, Bn, 64, HW);
  laplacian3x3<<<gridE((long)Bn * 64 * HW), blk, 0, stream>>>(XB, edge, Bn, 64, 256, 256);

  // 1x1 convs (per-b GEMM over pixels): r -> fin[0:32], e -> fin[32:64]
  gemm(P(105), XB, fin, 32, (int)HW, 64, 64, (int)HW, (int)HW, 1,
       0, (long)64 * HW, (long)64 * HW, 0, Bn, nullptr, 0);
  gemm(P(108), edge, fin + 32 * HW, 32, (int)HW, 64, 64, (int)HW, (int)HW, 1,
       0, (long)64 * HW, (long)64 * HW, 0, Bn, nullptr, 0);
  bnrelu<<<gridE((long)Bn * 32 * HW), blk, 0, stream>>>(fin, P(106), P(107), Bn, 64, 0, 32, HW);
  bnrelu<<<gridE((long)Bn * 32 * HW), blk, 0, stream>>>(fin, P(109), P(110), Bn, 64, 32, 32, HW);

  float* f = edge;   // edge no longer needed
  conv3x3_bnrelu<<<gridE((long)Bn * 64 * HW), blk, 0, stream>>>(
      fin, P(111), P(112), P(113), f, Bn, 64, 64, 256, 256);

  float* bam = Xf;   // Xf tokens no longer needed
  add2<<<gridE((long)Bn * 64 * HW), blk, 0, stream>>>(bam, XB, f, (long)Bn * 64 * HW);

  // seg: 1x1 conv 64->6 + bias, output NCHW (4,6,256,256)
  gemm(P(114), bam, (float*)d_out, 6, (int)HW, 64, 64, (int)HW, (int)HW, 1,
       0, (long)64 * HW, (long)6 * HW, 0, Bn, P(115), 2);
}
